// WalveAttentionPlainConvUNet_53291954209167
// MI455X (gfx1250) — matrix operand order repaired and checked by reading, not compiled, not run on hardware
//
#include <hip/hip_runtime.h>
#include <hip/hip_bf16.h>

typedef __attribute__((ext_vector_type(16))) _Float16 v16h;
typedef __attribute__((ext_vector_type(8)))  _Float16 v8h;
typedef __attribute__((ext_vector_type(8)))  float    v8f;

#define NSPAT 110592      // 48*48*48
#define XCH   884736      // 96*96*96

// ---------------------------------------------------------------------------
// Stage 0: zero the small accumulators in workspace
// ---------------------------------------------------------------------------
__global__ void init_ws(float* S, float* bnsums) {
    int i = threadIdx.x + blockIdx.x * 256;
    if (i < 512) S[i] = 0.0f;
    if (i < 64)  bnsums[i] = 0.0f;
}

// ---------------------------------------------------------------------------
// Stage A: parity sums S[b*32+c][p], p = pd*4+ph*2+pw (2x2x2 corner parity)
// One full-bandwidth pass over x; parities are static per thread by mapping
// w-parity to lane&1, h-parity to (tid/32)&1, and unrolling d in pairs.
// ---------------------------------------------------------------------------
__global__ __launch_bounds__(256) void parity_sums(const float* __restrict__ x,
                                                   float* __restrict__ S) {
    const int bc   = blockIdx.x;          // (b*32 + c), 0..63
    const int lane = threadIdx.x & 31;    // w base (parity = lane&1)
    const int hg   = threadIdx.x >> 5;    // h group 0..7 (parity = hg&1)
    const int d0   = blockIdx.y * 16;     // 6 slabs of 16 d-planes
    const float* xp = x + (size_t)bc * XCH;

    float acc0 = 0.0f;  // pd = 0
    float acc1 = 0.0f;  // pd = 1
    for (int dd = 0; dd < 8; ++dd) {
        const float* xd = xp + (size_t)(d0 + 2 * dd) * (96 * 96);
        for (int hh = 0; hh < 12; ++hh) {
            const float* xh = xd + (hg + 8 * hh) * 96;
            #pragma unroll
            for (int ww = 0; ww < 3; ++ww) {
                acc0 += xh[lane + 32 * ww];
                acc1 += xh[96 * 96 + lane + 32 * ww];
            }
        }
    }
    __shared__ float ps[8];
    if (threadIdx.x < 8) ps[threadIdx.x] = 0.0f;
    __syncthreads();
    const int ph = hg & 1, pw = lane & 1;
    atomicAdd(&ps[0 * 4 + ph * 2 + pw], acc0);
    atomicAdd(&ps[1 * 4 + ph * 2 + pw], acc1);
    __syncthreads();
    if (threadIdx.x < 8) atomicAdd(&S[bc * 8 + threadIdx.x], ps[threadIdx.x]);
}

// ---------------------------------------------------------------------------
// Stage B (tiny, 1 block): pooled means -> attention MLPs -> sigmoid scales ->
// fold scales + Haar transform into f16 GEMM weights w2h[b][o][c*8+corner].
// Band bitmask mb (D=4,H=2,W=1): 0=aaa(low), 1..7 = aad,ada,add,daa,dad,dda,ddd.
// sign(mb, p) = (-1)^popc(mb & p); amplitude (1/sqrt2)^3.
// ---------------------------------------------------------------------------
__global__ __launch_bounds__(256) void build_weights(
    const float* __restrict__ S,
    const float* __restrict__ w1_low,  const float* __restrict__ w2_low,
    const float* __restrict__ w1_high, const float* __restrict__ w2_high,
    const float* __restrict__ w_fuse,
    _Float16* __restrict__ w2h) {
    __shared__ float ylow[2][32], yhigh[2][224];
    __shared__ float hlow[2][2],  hhigh[2][14];
    __shared__ float slow[2][32], shigh[2][224];
    const float HAAR = 0.35355339059327373f;       // (1/sqrt2)^3
    const float K0 = HAAR / 110592.0f;             // * mean over 48^3
    const int t = threadIdx.x;

    // phase 1: sub-band spatial means from parity sums
    for (int it = t; it < 512; it += 256) {
        int b = it >> 8, idx = it & 255;
        if (idx < 32) {
            const float* sp = S + (b * 32 + idx) * 8;
            float a = 0.0f;
            for (int p = 0; p < 8; ++p) a += sp[p];
            ylow[b][idx] = a * K0;
        } else {
            int i = idx - 32;                     // 0..223 high channel
            int c = i / 7, mb = (i % 7) + 1;
            const float* sp = S + (b * 32 + c) * 8;
            float a = 0.0f;
            for (int p = 0; p < 8; ++p)
                a += ((__popc(mb & p) & 1) ? -sp[p] : sp[p]);
            yhigh[b][i] = a * K0;
        }
    }
    __syncthreads();
    // phase 2: hidden layers (ReLU)
    if (t < 4) {
        int b = t >> 1, j = t & 1;
        float a = 0.0f;
        for (int c = 0; c < 32; ++c) a += w1_low[j * 32 + c] * ylow[b][c];
        hlow[b][j] = fmaxf(a, 0.0f);
    } else if (t < 32) {
        int i2 = t - 4, b = i2 / 14, j = i2 % 14;
        float a = 0.0f;
        for (int i = 0; i < 224; ++i) a += w1_high[j * 224 + i] * yhigh[b][i];
        hhigh[b][j] = fmaxf(a, 0.0f);
    }
    __syncthreads();
    // phase 3: sigmoid scales
    for (int it = t; it < 512; it += 256) {
        int b = it >> 8, idx = it & 255;
        if (idx < 32) {
            float a = 0.0f;
            for (int j = 0; j < 2; ++j) a += w2_low[idx * 2 + j] * hlow[b][j];
            slow[b][idx] = 1.0f / (1.0f + expf(-a));
        } else {
            int i = idx - 32;
            float a = 0.0f;
            for (int j = 0; j < 14; ++j) a += w2_high[i * 14 + j] * hhigh[b][j];
            shigh[b][i] = 1.0f / (1.0f + expf(-a));
        }
    }
    __syncthreads();
    // phase 4: fold attention + Haar into f16 weights (K = c*8 + corner)
    for (int it = t; it < 2048; it += 256) {
        int b = it >> 10, oc = it & 1023, o = oc >> 5, c = oc & 31;
        float weff[8];
        weff[0] = w_fuse[o * 256 + c] * slow[b][c];
        #pragma unroll
        for (int mb = 1; mb < 8; ++mb)
            weff[mb] = w_fuse[o * 256 + 32 + c * 7 + (mb - 1)] * shigh[b][c * 7 + mb - 1];
        #pragma unroll
        for (int p = 0; p < 8; ++p) {
            float a = 0.0f;
            #pragma unroll
            for (int mb = 0; mb < 8; ++mb)
                a += ((__popc(mb & p) & 1) ? -weff[mb] : weff[mb]);
            w2h[(size_t)(b * 32 + o) * 256 + c * 8 + p] = (_Float16)(HAAR * a);
        }
    }
}

// ---------------------------------------------------------------------------
// Stage C: fused DWT+attention+1x1x1 conv as WMMA GEMM.
// Per wave: 16 half-res spatial columns x 32 output channels, K=256 in 8
// chunks of 32 using v_wmma_f32_16x16x32_f16 (two M-tiles per chunk).
// ---------------------------------------------------------------------------
__global__ __launch_bounds__(256) void fuse_gemm(const float* __restrict__ x,
                                                 const _Float16* __restrict__ w2h,
                                                 const float* __restrict__ b_fuse,
                                                 float* __restrict__ pre) {
    const int b    = blockIdx.y;
    const int wave = threadIdx.x >> 5;
    const int lane = threadIdx.x & 31;
    const int col0 = blockIdx.x * 128 + wave * 16;  // 48 = 3*16 -> wave stays in one w-row
    const int n    = lane & 15;                     // N position in tile
    const int hi   = lane >> 4;                     // lane half (K split)
    const int w0 = col0 % 48;
    const int h  = (col0 / 48) % 48;
    const int d  = col0 / (48 * 48);

    const float*    xb = x + (size_t)b * 32 * XCH
                           + (size_t)(2 * d) * (96 * 96) + (2 * h) * 96 + 2 * w0 + 2 * n;
    const _Float16* wb = w2h + (size_t)b * 32 * 256;

    v8f c0 = {}; v8f c1 = {};
    #pragma unroll
    for (int kc = 0; kc < 8; ++kc) {
        const int kb = kc * 32;
        // ---- B fragment: K = kb + hi*16 + e ; e = (c-cbase)*8 + pd*4 + ph*2 + pw
        v16h bf;
        const int cbase = 4 * kc + 2 * hi;
        #pragma unroll
        for (int cc = 0; cc < 2; ++cc) {
            const float* xc = xb + (size_t)(cbase + cc) * XCH;
            #pragma unroll
            for (int pd = 0; pd < 2; ++pd) {
                #pragma unroll
                for (int ph = 0; ph < 2; ++ph) {
                    const float2 v = *(const float2*)(xc + pd * (96 * 96) + ph * 96);
                    const int e = cc * 8 + pd * 4 + ph * 2;
                    bf[e]     = (_Float16)v.x;
                    bf[e + 1] = (_Float16)v.y;
                }
            }
        }
        // ---- A fragments (rows = output channels): lane half -> K = kb+hi*16 .. kb+hi*16+15
        {
            const _Float16* ar = wb + (size_t)n * 256 + kb + hi * 16;
            v8h alo = *(const v8h*)(ar);
            v8h ahi = *(const v8h*)(ar + 8);
            v16h af;
            #pragma unroll
            for (int i = 0; i < 8; ++i) { af[i] = alo[i]; af[8 + i] = ahi[i]; }
            c0 = __builtin_amdgcn_wmma_f32_16x16x32_f16(false, af, false, bf,
                                                        (short)0, c0, false, false);
        }
        {
            const _Float16* ar = wb + (size_t)(16 + n) * 256 + kb + hi * 16;
            v8h alo = *(const v8h*)(ar);
            v8h ahi = *(const v8h*)(ar + 8);
            v16h af;
            #pragma unroll
            for (int i = 0; i < 8; ++i) { af[i] = alo[i]; af[8 + i] = ahi[i]; }
            c1 = __builtin_amdgcn_wmma_f32_16x16x32_f16(false, af, false, bf,
                                                        (short)0, c1, false, false);
        }
    }
    // ---- store with bias; C/D layout: VGPR r, lane half -> M = r + hi*8
    float* ob = pre + (size_t)b * 32 * NSPAT + col0 + n;
    #pragma unroll
    for (int r = 0; r < 8; ++r) {
        const int o0 = r + hi * 8;
        ob[(size_t)o0 * NSPAT]        = c0[r] + b_fuse[o0];
        ob[(size_t)(16 + o0) * NSPAT] = c1[r] + b_fuse[16 + o0];
    }
}

// ---------------------------------------------------------------------------
// Stage D: per-output-channel sum / sumsq for BatchNorm (over B and spatial)
// ---------------------------------------------------------------------------
__global__ __launch_bounds__(256) void bn_stats(const float* __restrict__ pre,
                                                float* __restrict__ sums) {
    const int o = blockIdx.x, part = blockIdx.y;   // 32 x 8 blocks
    const int chunk = NSPAT / 8;
    float s = 0.0f, q = 0.0f;
    for (int b = 0; b < 2; ++b) {
        const float* p = pre + (size_t)b * 32 * NSPAT + (size_t)o * NSPAT;
        for (int i = part * chunk + threadIdx.x; i < (part + 1) * chunk; i += 256) {
            float v = p[i];
            s += v; q += v * v;
        }
    }
    __shared__ float ls[256], lq[256];
    ls[threadIdx.x] = s; lq[threadIdx.x] = q;
    __syncthreads();
    for (int off = 128; off; off >>= 1) {
        if (threadIdx.x < off) {
            ls[threadIdx.x] += ls[threadIdx.x + off];
            lq[threadIdx.x] += lq[threadIdx.x + off];
        }
        __syncthreads();
    }
    if (threadIdx.x == 0) {
        atomicAdd(&sums[o], ls[0]);
        atomicAdd(&sums[32 + o], lq[0]);
    }
}

__global__ void bn_finalize(const float* __restrict__ sums,
                            const float* __restrict__ gamma,
                            const float* __restrict__ beta,
                            float* __restrict__ ss) {
    int o = threadIdx.x;
    if (o < 32) {
        const float N  = 2.0f * NSPAT;
        float mu  = sums[o] / N;
        float var = sums[32 + o] / N - mu * mu;
        float inv = rsqrtf(var + 1e-5f);
        float sc  = gamma[o] * inv;
        ss[o]      = sc;
        ss[32 + o] = beta[o] - mu * sc;
    }
}

// ---------------------------------------------------------------------------
// Stage E: z*scale + shift, ReLU  (float4 vectorized; channel stride is /4)
// ---------------------------------------------------------------------------
__global__ __launch_bounds__(1024) void bn_apply(const float* __restrict__ pre,
                                                 const float* __restrict__ ss,
                                                 float* __restrict__ out) {
    const size_t i = (size_t)blockIdx.x * 1024 + threadIdx.x;  // float4 index
    const float4 v = ((const float4*)pre)[i];
    const int o = (int)(((i * 4) / NSPAT) & 31);
    const float sc = ss[o], sh = ss[32 + o];
    float4 r;
    r.x = fmaxf(v.x * sc + sh, 0.0f);
    r.y = fmaxf(v.y * sc + sh, 0.0f);
    r.z = fmaxf(v.z * sc + sh, 0.0f);
    r.w = fmaxf(v.w * sc + sh, 0.0f);
    ((float4*)out)[i] = r;
}

// ---------------------------------------------------------------------------
extern "C" void kernel_launch(void* const* d_in, const int* in_sizes, int n_in,
                              void* d_out, int out_size, void* d_ws, size_t ws_size,
                              hipStream_t stream) {
    (void)in_sizes; (void)n_in; (void)out_size; (void)ws_size;
    const float* x       = (const float*)d_in[0];
    const float* w1_low  = (const float*)d_in[1];
    const float* w2_low  = (const float*)d_in[2];
    const float* w1_high = (const float*)d_in[3];
    const float* w2_high = (const float*)d_in[4];
    const float* w_fuse  = (const float*)d_in[5];
    const float* b_fuse  = (const float*)d_in[6];
    const float* gamma   = (const float*)d_in[7];
    const float* beta    = (const float*)d_in[8];

    char* ws = (char*)d_ws;
    float*     S       = (float*)(ws + 0);        // 512 f32
    float*     bnsums  = (float*)(ws + 2048);     // 64 f32
    float*     bnscale = (float*)(ws + 2304);     // 64 f32
    _Float16*  w2h     = (_Float16*)(ws + 4096);  // 2*32*256 f16 (32 KB)
    float*     pre     = (float*)(ws + 65536);    // 2*32*110592 f32 (~27 MB)

    init_ws<<<2, 256, 0, stream>>>(S, bnsums);
    parity_sums<<<dim3(64, 6), 256, 0, stream>>>(x, S);
    build_weights<<<1, 256, 0, stream>>>(S, w1_low, w2_low, w1_high, w2_high,
                                         w_fuse, w2h);
    fuse_gemm<<<dim3(864, 2), 256, 0, stream>>>(x, w2h, b_fuse, pre);
    bn_stats<<<dim3(32, 8), 256, 0, stream>>>(pre, bnsums);
    bn_finalize<<<1, 64, 0, stream>>>(bnsums, gamma, beta, bnscale);
    bn_apply<<<1728, 1024, 0, stream>>>(pre, bnscale, (float*)d_out);
}
